// TwoStreamDynamicBlock_78503412236442
// MI455X (gfx1250) — matrix operand order repaired and checked by reading, not compiled, not run on hardware
//
#include <hip/hip_runtime.h>
#include <hip/hip_bf16.h>

typedef __attribute__((ext_vector_type(16))) _Float16 v16h;
typedef __attribute__((ext_vector_type(8)))  float    v8f;
typedef __attribute__((ext_vector_type(4)))  unsigned int u32x4;
typedef __attribute__((ext_vector_type(8)))  unsigned int u32x8;

#define BB 16
#define NN 2048
#define CC 64
#define HH 128
#define KKN 16
#define NTOT (BB*NN)        /* 32768 nodes  */
#define R1   (NTOT*KKN)     /* 524288 edge rows */
#define BNEPS 1e-5f
#define AHP 264             /* gemmF LDS A-tile row stride in halves (512B + 16B TDM pad) */

// ---------------------------------------------------------------- utilities
__global__ void zero_kernel(float* p, int n) {
    int i = blockIdx.x * blockDim.x + threadIdx.x;
    if (i < n) p[i] = 0.f;
}

__global__ void sqnorm_kernel(const float* __restrict__ x, float* __restrict__ sq) {
    int n = blockIdx.x * blockDim.x + threadIdx.x;
    if (n >= NTOT) return;
    const float* r = x + (size_t)n * CC;
    float s = 0.f;
#pragma unroll
    for (int d = 0; d < CC; ++d) s += r[d] * r[d];
    sq[n] = s;
}

// ---------------------------------------------------------------- kNN (shared by both streams)
// grid: (NN/4, BB), block: 128 (4 waves, one row per wave)
__global__ void knn_kernel(const float* __restrict__ x, const float* __restrict__ sq,
                           int* __restrict__ idxw) {
    __shared__ float xm[64][65];     // 64-node tile, padded vs bank conflicts
    __shared__ float xn[4][64];
    __shared__ float d2s[4][2048];

    int b = blockIdx.y, r0 = blockIdx.x * 4;
    int tid = threadIdx.x, w = tid >> 5, lane = tid & 31;
    const float* xb = x + (size_t)b * NN * CC;

    for (int i = tid; i < 4 * 64; i += 128)
        xn[i >> 6][i & 63] = xb[(size_t)(r0 + (i >> 6)) * CC + (i & 63)];
    float sqn = sq[b * NN + r0 + w];

    for (int t = 0; t < NN / 64; ++t) {
        int m0 = t * 64;
        __syncthreads();
        for (int i = tid; i < 64 * 64; i += 128)
            xm[i >> 6][i & 63] = xb[(size_t)(m0 + (i >> 6)) * CC + (i & 63)];
        __syncthreads();
        for (int mm = lane; mm < 64; mm += 32) {
            float dot = 0.f;
#pragma unroll
            for (int d = 0; d < CC; ++d) dot += xn[w][d] * xm[mm][d];
            int m = m0 + mm;
            d2s[w][m] = sqn + sq[b * NN + m] - 2.f * dot;
        }
    }
    __syncthreads();

    // iterative top-16 smallest (stable: smaller index wins ties)
    int* idxRow = idxw + (size_t)(b * NN + r0 + w) * KKN;
    for (int sel = 0; sel < KKN; ++sel) {
        float bv = 3.4e38f; int bi = 0x7fffffff;
        for (int i = lane; i < NN; i += 32) {
            float v = d2s[w][i];
            if (v < bv) { bv = v; bi = i; }
        }
#pragma unroll
        for (int off = 16; off > 0; off >>= 1) {
            float ov = __shfl_down(bv, off, 32);
            int   oi = __shfl_down(bi, off, 32);
            if (ov < bv || (ov == bv && oi < bi)) { bv = ov; bi = oi; }
        }
        bi = __shfl(bi, 0, 32);
        if (lane == 0) { idxRow[sel] = bi; d2s[w][bi] = 3.4e38f; }
    }
}

// ---------------------------------------------------------------- P = x@(W1a-W1b), Q = x@W1b
// grid: NTOT, block: 128 (thread = output channel)
__global__ void pq_kernel(const float* __restrict__ x, const float* __restrict__ W1,
                          float* __restrict__ P, float* __restrict__ Q) {
    __shared__ float xr[CC];
    int n = blockIdx.x, c = threadIdx.x;
    if (c < CC) xr[c] = x[(size_t)n * CC + c];
    __syncthreads();
    float accP = 0.f, accQ = 0.f;
#pragma unroll 8
    for (int d = 0; d < CC; ++d) {
        float wa = W1[d * HH + c];
        float wb = W1[(CC + d) * HH + c];
        accP += xr[d] * (wa - wb);
        accQ += xr[d] * wb;
    }
    P[(size_t)n * HH + c] = accP;
    Q[(size_t)n * HH + c] = accQ;
}

// ---------------------------------------------------------------- BN1 stats over edge rows
// grid: R1/64, block: 128 (thread = channel)
__global__ void stats1_kernel(const float* __restrict__ P, const float* __restrict__ Q,
                              const int* __restrict__ idxw, const float* __restrict__ b1,
                              float* __restrict__ sum, float* __restrict__ sumsq) {
    int c = threadIdx.x;
    long base = (long)blockIdx.x * 64;
    float bc = b1[c], s = 0.f, q = 0.f;
    for (int rr = 0; rr < 64; ++rr) {
        long r = base + rr;
        int n = (int)(r >> 4);
        int g = ((n >> 11) << 11) + idxw[r];
        float v = P[(size_t)n * HH + c] + Q[(size_t)g * HH + c] + bc;
        s += v; q += v * v;
    }
    atomicAdd(&sum[c], s);
    atomicAdd(&sumsq[c], q);
}

// ---------------------------------------------------------------- BN finalize: (sum,sumsq)->(scale,shift)
__global__ void finalize_bn_kernel(const float* __restrict__ sum, const float* __restrict__ sumsq,
                                   float count, const float* __restrict__ g,
                                   const float* __restrict__ be,
                                   float* __restrict__ scale, float* __restrict__ shift) {
    int c = threadIdx.x;
    float m = sum[c] / count;
    float v = sumsq[c] / count - m * m;
    float s = g[c] * rsqrtf(v + BNEPS);
    scale[c] = s;
    shift[c] = be[c] - m * s;
}

// ---------------------------------------------------------------- h2 = relu(bn1(h1)) @ W2 + b2 via WMMA
// Each 16-row WMMA tile == one node's K=16 neighbors -> max/min over K inside the tile.
// grid: NTOT/8, block: 256 (8 waves; wave w owns column tile w*16)
__global__ void gemm2_kernel(const float* __restrict__ P, const float* __restrict__ Q,
                             const int* __restrict__ idxw, const float* __restrict__ b1,
                             const float* __restrict__ sc1, const float* __restrict__ sh1,
                             const float* __restrict__ W2, const float* __restrict__ b2,
                             float* __restrict__ mx, float* __restrict__ mn,
                             float* __restrict__ sum2, float* __restrict__ sumsq2) {
    __shared__ _Float16 w2h[HH * HH];       // 32 KB
    __shared__ _Float16 a1h[16 * 130];      // padded (odd word stride)

    int tid = threadIdx.x, lane = tid & 31, wv = tid >> 5;
    int half = lane >> 4, l16 = lane & 15;
    int n0 = wv * 16;

    for (int i = tid; i < HH * HH; i += 256) w2h[i] = (_Float16)W2[i];
    __syncthreads();

    // B fragments: VGPR v, lanes0-15 K={2v,2v+1}; lanes16-31 K={16+2v,16+2v+1}
    v16h bfrag[4];
#pragma unroll
    for (int s = 0; s < 4; ++s) {
#pragma unroll
        for (int v = 0; v < 8; ++v) {
            int k = 32 * s + 2 * v + 16 * half;
            bfrag[s][2 * v]     = w2h[k * HH + n0 + l16];
            bfrag[s][2 * v + 1] = w2h[(k + 1) * HH + n0 + l16];
        }
    }
    float b2c = b2[n0 + l16];
    float regS = 0.f, regQ = 0.f;

    int nodeBase = blockIdx.x * 8;
    for (int nl = 0; nl < 8; ++nl) {
        int n = nodeBase + nl;
        int gb = (n >> 11) << 11;
        __syncthreads();
        {   // stage a1 = relu(bn1(P[n] + Q[idx] + b1)) as f16, 16x128
            int r = tid >> 4, c0 = (tid & 15) * 8;
            int g = gb + idxw[(size_t)n * KKN + r];
            const float* Pp = P + (size_t)n * HH;
            const float* Qp = Q + (size_t)g * HH;
#pragma unroll
            for (int cc = 0; cc < 8; ++cc) {
                int c = c0 + cc;
                float v = Pp[c] + Qp[c] + b1[c];
                v = v * sc1[c] + sh1[c];
                a1h[r * 130 + c] = (_Float16)(v > 0.f ? v : 0.f);
            }
        }
        __syncthreads();

        v8f acc = {};
#pragma unroll
        for (int s = 0; s < 4; ++s) {
            v16h af;
#pragma unroll
            for (int p = 0; p < 8; ++p) {
                int koff = 2 * p + 16 * half;  // contiguous K order, matches B fragments
                af[2 * p]     = a1h[l16 * 130 + 32 * s + koff];
                af[2 * p + 1] = a1h[l16 * 130 + 32 * s + koff + 1];
            }
            acc = __builtin_amdgcn_wmma_f32_16x16x32_f16(false, af, false, bfrag[s],
                                                         (short)0, acc, false, false);
        }
        // epilogue: bias, max/min over the 16 K-rows, BN2 stat partials
        float vmax = -3.4e38f, vmin = 3.4e38f;
#pragma unroll
        for (int r = 0; r < 8; ++r) {
            float h = acc[r] + b2c;
            vmax = fmaxf(vmax, h); vmin = fminf(vmin, h);
            regS += h; regQ += h * h;
        }
        vmax = fmaxf(vmax, __shfl_xor(vmax, 16, 32));
        vmin = fminf(vmin, __shfl_xor(vmin, 16, 32));
        if (half == 0) {
            mx[(size_t)n * HH + n0 + l16] = vmax;
            mn[(size_t)n * HH + n0 + l16] = vmin;
        }
    }
    atomicAdd(&sum2[n0 + l16], regS);
    atomicAdd(&sumsq2[n0 + l16], regQ);
}

// ---------------------------------------------------------------- xs = relu(bn2(max/min)) -> comb (f16)
__global__ void combine_kernel(const float* __restrict__ mx, const float* __restrict__ mn,
                               const float* __restrict__ sc2, const float* __restrict__ sh2,
                               _Float16* __restrict__ comb, int colOff) {
    int i = blockIdx.x * blockDim.x + threadIdx.x;   // over NTOT*HH
    if (i >= NTOT * HH) return;
    int n = i >> 7, c = i & 127;
    float s = sc2[c];
    float h = (s >= 0.f ? mx[i] : mn[i]) * s + sh2[c];   // relu(s*h+t) extremum selection
    comb[(size_t)n * (2 * HH) + colOff + c] = (_Float16)(h > 0.f ? h : 0.f);
}

// ---------------------------------------------------------------- hF = comb @ fW + fb via WMMA + TDM
// grid: NTOT/128, block: 256 (8 waves; wave = 16-col tile; 8 row-tiles of 16)
// A-tiles (16 x 512B rows of comb) are fetched by the Tensor Data Mover with
// pad_enable adding 16B per 512B row -> LDS row stride 264 halves.
__global__ void gemmF_kernel(const _Float16* __restrict__ comb, const float* __restrict__ fW,
                             const float* __restrict__ fb, float* __restrict__ hF,
                             float* __restrict__ sumF, float* __restrict__ sumsqF) {
    __shared__ _Float16 ah[16 * AHP];   // 16 x 256 + 8 pad per row (TDM-written)
    int tid = threadIdx.x, lane = tid & 31, wv = tid >> 5;
    int half = lane >> 4, l16 = lane & 15;
    int n0 = wv * 16;

    v16h bfrag[8];
#pragma unroll
    for (int s = 0; s < 8; ++s) {
#pragma unroll
        for (int v = 0; v < 8; ++v) {
            int k = 32 * s + 2 * v + 16 * half;
            bfrag[s][2 * v]     = (_Float16)fW[k * HH + n0 + l16];
            bfrag[s][2 * v + 1] = (_Float16)fW[(k + 1) * HH + n0 + l16];
        }
    }
    float fbc = fb[n0 + l16];
    float regS = 0.f, regQ = 0.f;
    int rowBase0 = blockIdx.x * 128;
    unsigned ldsAddr = (unsigned)(size_t)&ah[0];   // addr[31:0] == LDS byte offset

    for (int mt = 0; mt < 8; ++mt) {
        int rb = rowBase0 + mt * 16;
        __syncthreads();            // previous tile fully consumed before overwrite
        if (wv == 0) {
            // ---- Tensor DMA descriptor (D#), groups 0 & 1 (2D tile) ----
            unsigned long long gaddr =
                (unsigned long long)(size_t)(comb + (size_t)rb * (2 * HH));
            u32x4 g0;
            g0[0] = 1u;                                    // count=1, user mode
            g0[1] = ldsAddr;                               // lds_addr
            g0[2] = (unsigned)gaddr;                       // global_addr[31:0]
            g0[3] = (unsigned)((gaddr >> 32) & 0x01FFFFFFull) | (2u << 30); // [56:32] | type=2
            u32x8 g1;
            // data_size=3 (8B), pad_enable, pad_interval=6 (128 DW = 512B), pad_amount=3 (4 DW = 16B)
            g1[0] = (3u << 16) | (1u << 20) | (6u << 22) | (3u << 25);
            g1[1] = (64u << 16);                           // tensor_dim0 = 64 units of 8B (512B row)
            g1[2] = (16u << 16);                           // tensor_dim1 = 16 rows
            g1[3] = (64u << 16);                           // tile_dim0 = 64
            g1[4] = 16u;                                   // tile_dim1 = 16, tile_dim2 = 0
            g1[5] = 64u;                                   // tensor_dim0_stride = 64 units (contiguous)
            g1[6] = 0u;
            g1[7] = 0u;
            asm volatile("tensor_load_to_lds %0, %1" :: "s"(g0), "s"(g1) : "memory");
            __builtin_amdgcn_s_wait_tensorcnt(0);
        }
        __syncthreads();            // publish TDM-written LDS tile to all waves

        v8f acc = {};
#pragma unroll
        for (int s = 0; s < 8; ++s) {
            v16h af;
#pragma unroll
            for (int p = 0; p < 8; ++p) {
                int koff = 2 * p + 16 * half;
                af[2 * p]     = ah[l16 * AHP + 32 * s + koff];
                af[2 * p + 1] = ah[l16 * AHP + 32 * s + koff + 1];
            }
            acc = __builtin_amdgcn_wmma_f32_16x16x32_f16(false, af, false, bfrag[s],
                                                         (short)0, acc, false, false);
        }
#pragma unroll
        for (int r = 0; r < 8; ++r) {
            float h = acc[r] + fbc;
            hF[(size_t)(rb + r + 8 * half) * HH + n0 + l16] = h;
            regS += h; regQ += h * h;
        }
    }
    atomicAdd(&sumF[n0 + l16], regS);
    atomicAdd(&sumsqF[n0 + l16], regQ);
}

// ---------------------------------------------------------------- out = relu(bnF(hF))
__global__ void outF_kernel(const float* __restrict__ hF, const float* __restrict__ scF,
                            const float* __restrict__ shF, float* __restrict__ out) {
    int i = blockIdx.x * blockDim.x + threadIdx.x;
    if (i >= NTOT * HH) return;
    int c = i & 127;
    float h = hF[i] * scF[c] + shF[c];
    out[i] = h > 0.f ? h : 0.f;
}

// ================================================================ host
extern "C" void kernel_launch(void* const* d_in, const int* in_sizes, int n_in,
                              void* d_out, int out_size, void* d_ws, size_t ws_size,
                              hipStream_t stream) {
    const float* x    = (const float*)d_in[0];
    // d_in[1] = batch (structure known: B contiguous graphs of N) -> unused
    const float* sW1  = (const float*)d_in[2];
    const float* sb1  = (const float*)d_in[3];
    const float* sg1  = (const float*)d_in[4];
    const float* sbe1 = (const float*)d_in[5];
    const float* sW2  = (const float*)d_in[6];
    const float* sb2  = (const float*)d_in[7];
    const float* sg2  = (const float*)d_in[8];
    const float* sbe2 = (const float*)d_in[9];
    const float* tW1  = (const float*)d_in[10];
    const float* tb1  = (const float*)d_in[11];
    const float* tg1  = (const float*)d_in[12];
    const float* tbe1 = (const float*)d_in[13];
    const float* tW2  = (const float*)d_in[14];
    const float* tb2  = (const float*)d_in[15];
    const float* tg2  = (const float*)d_in[16];
    const float* tbe2 = (const float*)d_in[17];
    const float* fW   = (const float*)d_in[18];
    const float* fb   = (const float*)d_in[19];
    const float* fg   = (const float*)d_in[20];
    const float* fbe  = (const float*)d_in[21];
    float* out = (float*)d_out;

    // workspace carving (256B aligned)
    char* p = (char*)d_ws;
    auto carve = [&](size_t bytes) { void* r = (void*)p; p += (bytes + 255) & ~(size_t)255; return r; };
    float*     sq    = (float*)carve((size_t)NTOT * 4);
    int*       idxw  = (int*)carve((size_t)R1 * 4);
    float*     P     = (float*)carve((size_t)NTOT * HH * 4);
    float*     Q     = (float*)carve((size_t)NTOT * HH * 4);
    float*     mx    = (float*)carve((size_t)NTOT * HH * 4);
    float*     mn    = (float*)carve((size_t)NTOT * HH * 4);
    _Float16*  comb  = (_Float16*)carve((size_t)NTOT * 2 * HH * 2);
    float*     hF    = (float*)carve((size_t)NTOT * HH * 4);
    float*     stats = (float*)carve(1280 * 4);   // 5 BN stages x (sum,sumsq) x 128
    float*     bnp   = (float*)carve(1280 * 4);   // 5 BN stages x (scale,shift) x 128

    float *st1_sum = stats,        *st1_sq = stats + 128;
    float *st2_sum = stats + 256,  *st2_sq = stats + 384;
    float *st3_sum = stats + 512,  *st3_sq = stats + 640;
    float *st4_sum = stats + 768,  *st4_sq = stats + 896;
    float *stF_sum = stats + 1024, *stF_sq = stats + 1152;
    float *bn1s_s = bnp,        *bn1s_t = bnp + 128;
    float *bn2s_s = bnp + 256,  *bn2s_t = bnp + 384;
    float *bn1t_s = bnp + 512,  *bn1t_t = bnp + 640;
    float *bn2t_s = bnp + 768,  *bn2t_t = bnp + 896;
    float *bnF_s  = bnp + 1024, *bnF_t  = bnp + 1152;

    zero_kernel<<<(1280 + 255) / 256, 256, 0, stream>>>(stats, 1280);
    sqnorm_kernel<<<NTOT / 256, 256, 0, stream>>>(x, sq);
    knn_kernel<<<dim3(NN / 4, BB), 128, 0, stream>>>(x, sq, idxw);

    // ---- stream s ----
    pq_kernel<<<NTOT, 128, 0, stream>>>(x, sW1, P, Q);
    stats1_kernel<<<R1 / 64, 128, 0, stream>>>(P, Q, idxw, sb1, st1_sum, st1_sq);
    finalize_bn_kernel<<<1, 128, 0, stream>>>(st1_sum, st1_sq, (float)R1, sg1, sbe1, bn1s_s, bn1s_t);
    gemm2_kernel<<<NTOT / 8, 256, 0, stream>>>(P, Q, idxw, sb1, bn1s_s, bn1s_t, sW2, sb2,
                                               mx, mn, st2_sum, st2_sq);
    finalize_bn_kernel<<<1, 128, 0, stream>>>(st2_sum, st2_sq, (float)R1, sg2, sbe2, bn2s_s, bn2s_t);
    combine_kernel<<<(NTOT * HH) / 256, 256, 0, stream>>>(mx, mn, bn2s_s, bn2s_t, comb, 0);

    // ---- stream t (reuses P/Q/mx/mn) ----
    pq_kernel<<<NTOT, 128, 0, stream>>>(x, tW1, P, Q);
    stats1_kernel<<<R1 / 64, 128, 0, stream>>>(P, Q, idxw, tb1, st3_sum, st3_sq);
    finalize_bn_kernel<<<1, 128, 0, stream>>>(st3_sum, st3_sq, (float)R1, tg1, tbe1, bn1t_s, bn1t_t);
    gemm2_kernel<<<NTOT / 8, 256, 0, stream>>>(P, Q, idxw, tb1, bn1t_s, bn1t_t, tW2, tb2,
                                               mx, mn, st4_sum, st4_sq);
    finalize_bn_kernel<<<1, 128, 0, stream>>>(st4_sum, st4_sq, (float)R1, tg2, tbe2, bn2t_s, bn2t_t);
    combine_kernel<<<(NTOT * HH) / 256, 256, 0, stream>>>(mx, mn, bn2t_s, bn2t_t, comb, HH);

    // ---- fused head ----
    gemmF_kernel<<<NTOT / 128, 256, 0, stream>>>(comb, fW, fb, hF, stF_sum, stF_sq);
    finalize_bn_kernel<<<1, 128, 0, stream>>>(stF_sum, stF_sq, (float)NTOT, fg, fbe, bnF_s, bnF_t);
    outF_kernel<<<(NTOT * HH) / 256, 256, 0, stream>>>(hF, bnF_s, bnF_t, out);
}
